// GCN_44023414784008
// MI455X (gfx1250) — matrix operand order repaired and checked by reading, not compiled, not run on hardware
//
#include <hip/hip_runtime.h>

typedef __attribute__((ext_vector_type(8)))  unsigned short v8us;
typedef __attribute__((ext_vector_type(16))) unsigned short v16us;
typedef __attribute__((ext_vector_type(16))) __bf16         v16bf;
typedef __attribute__((ext_vector_type(8)))  float          v8f;

static __device__ __forceinline__ unsigned short f32_to_bf16(float f) {
  union { float f; unsigned u; } v;
  v.f = f;
  unsigned r = (v.u + 0x7FFFu + ((v.u >> 16) & 1u)) >> 16;  // RNE
  return (unsigned short)r;
}

static __device__ __forceinline__ v16bf mk_frag(v8us lo, v8us hi) {
  v16us t = __builtin_shufflevector(lo, hi, 0, 1, 2, 3, 4, 5, 6, 7, 8, 9, 10,
                                    11, 12, 13, 14, 15);
  return __builtin_bit_cast(v16bf, t);
}

static __device__ __forceinline__ v8f wmma_bf16(v16bf a, v16bf b, v8f c) {
  return __builtin_amdgcn_wmma_f32_16x16x32_bf16(false, a, false, b, (short)0,
                                                 c, false, false);
}

// Async DMA: global (16B per lane) -> LDS, tracked by ASYNCcnt.
static __device__ __forceinline__ void async_ld_b128(unsigned ldsOff,
                                                     const unsigned short* g) {
  asm volatile("global_load_async_to_lds_b128 %0, %1, off" ::"v"(ldsOff),
               "v"(g)
               : "memory");
}

// ---------------------------------------------------------------------------
// Elementwise fp32 -> bf16 (8 elements / thread)
// ---------------------------------------------------------------------------
__global__ __launch_bounds__(256) void cvt_bf16(const float* __restrict__ src,
                                                unsigned short* __restrict__ dst) {
  const size_t i = (size_t)blockIdx.x * 256 + threadIdx.x;
  const float4* s = (const float4*)src + i * 2;
  float4 v0 = s[0], v1 = s[1];
  v8us o;
  o[0] = f32_to_bf16(v0.x); o[1] = f32_to_bf16(v0.y);
  o[2] = f32_to_bf16(v0.z); o[3] = f32_to_bf16(v0.w);
  o[4] = f32_to_bf16(v1.x); o[5] = f32_to_bf16(v1.y);
  o[6] = f32_to_bf16(v1.z); o[7] = f32_to_bf16(v1.w);
  ((v8us*)dst)[i] = o;
}

// ---------------------------------------------------------------------------
// Per-batch transpose+convert: H (2048k x 1024c f32) -> Ht (1024c x 2048k bf16)
// ---------------------------------------------------------------------------
__global__ __launch_bounds__(256) void transpose_h_bf16(
    const float* __restrict__ H, unsigned short* __restrict__ Ht) {
  constexpr int K = 2048, C = 1024, TK = 64, TC = 64;
  constexpr int TPB = (K / TK) * (C / TC);  // 512 tiles per batch
  const int tile = blockIdx.x;
  const int b  = tile / TPB;
  const int t  = tile % TPB;
  const int tk = t / (C / TC);
  const int tc = t % (C / TC);
  const float* Hb = H + (size_t)b * K * C + (size_t)tk * TK * C + tc * TC;
  unsigned short* Tb = Ht + (size_t)b * C * K + (size_t)tc * TC * K + tk * TK;

  __shared__ __align__(16) unsigned short lds[TC][TK + 16];

  const int tid = threadIdx.x;
  const int cc = (tid & 15) * 4;
  const int rr = tid >> 4;
#pragma unroll
  for (int r = rr; r < TK; r += 16) {
    float4 v = *(const float4*)(Hb + (size_t)r * C + cc);
    lds[cc + 0][r] = f32_to_bf16(v.x);
    lds[cc + 1][r] = f32_to_bf16(v.y);
    lds[cc + 2][r] = f32_to_bf16(v.z);
    lds[cc + 3][r] = f32_to_bf16(v.w);
  }
  __syncthreads();
  const int oc = tid >> 2;
  const int ok = (tid & 3) * 16;
  v8us x0 = *(const v8us*)&lds[oc][ok];
  v8us x1 = *(const v8us*)&lds[oc][ok + 8];
  *(v8us*)(Tb + (size_t)oc * K + ok)     = x0;
  *(v8us*)(Tb + (size_t)oc * K + ok + 8) = x1;
}

// ---------------------------------------------------------------------------
// GEMM1: per batch, AH(2048x1024 bf16) = A(2048x2048 bf16) @ H(2048x1024)
// Block tile 256x64 (8 waves as 4x2, each wave 64x32 = 4x2 fragments).
// K staged 32-deep through LDS with double-buffered async global->LDS DMA.
// LDS rows padded to 80B: conflict-free ds_load_b128 fragment reads.
// ---------------------------------------------------------------------------
__global__ __launch_bounds__(256) void gcn_aggregate_bf16(
    const unsigned short* __restrict__ Abf,
    const unsigned short* __restrict__ Htb, unsigned short* __restrict__ AH) {
  constexpr int N = 2048, K = 2048, C = 1024;
  constexpr int BPB = (N / 256) * (C / 64);  // 8*16 = 128 blocks per batch
  constexpr int NK = K / 32;                 // 64 k-steps
  constexpr int ROWP = 40;                   // LDS row pitch in ushorts (80B)

  const int b   = blockIdx.x / BPB;
  const int t   = blockIdx.x % BPB;
  const int bm  = t / (C / 64);   // block row-tile (256 rows)
  const int bn  = t % (C / 64);   // block col-tile (64 cols)
  const int tid = threadIdx.x;
  const int wave = tid >> 5;
  const int lane = tid & 31;
  const int wr = wave >> 1;       // 0..3 : 64-row slice
  const int wc = wave & 1;        // 0..1 : 32-col slice
  const int half = lane >> 4;
  const int l15  = lane & 15;

  const unsigned short* Ab = Abf + (size_t)b * N * K + (size_t)bm * 256 * K;
  const unsigned short* Tb = Htb + (size_t)b * C * K + (size_t)bn * 64 * K;
  unsigned short* Ob       = AH + (size_t)b * N * C;

  // [2 buffers][256 rows A | 64 rows B] each row 40 ushorts (80 bytes)
  __shared__ __align__(16) unsigned short sA[2][256 * ROWP];
  __shared__ __align__(16) unsigned short sB[2][64 * ROWP];

  const unsigned aBase0 = (unsigned)(uintptr_t)&sA[0][0];
  const unsigned aBase1 = (unsigned)(uintptr_t)&sA[1][0];
  const unsigned bBase0 = (unsigned)(uintptr_t)&sB[0][0];
  const unsigned bBase1 = (unsigned)(uintptr_t)&sB[1][0];

  // Stage one 32-deep k-slab: A 256x32 (16KB) + B 64x32 (4KB); 5 b128/thread.
  const int sr = tid >> 2;           // 0..63
  const int kc = tid & 3;            // 16B chunk within a 64B row segment
  auto stage = [&](int k0, unsigned aBase, unsigned bBase) {
#pragma unroll
    for (int rr = 0; rr < 4; ++rr) {
      const int row = rr * 64 + sr;
      async_ld_b128(aBase + row * 80 + kc * 16, Ab + (size_t)row * K + k0 + kc * 8);
    }
    async_ld_b128(bBase + sr * 80 + kc * 16, Tb + (size_t)sr * K + k0 + kc * 8);
  };

  v8f acc[4][2] = {};
  stage(0, aBase0, bBase0);

  for (int i = 0; i < NK; ++i) {
    const int cur = i & 1;
    if (i + 1 < NK) {
      stage((i + 1) * 32, cur ? aBase0 : aBase1, cur ? bBase0 : bBase1);
      asm volatile("s_wait_asynccnt 0x5" ::: "memory");  // prev slab complete
    } else {
      asm volatile("s_wait_asynccnt 0x0" ::: "memory");
    }
    __syncthreads();

    const unsigned short* a_lds = sA[cur];
    const unsigned short* b_lds = sB[cur];
    v16bf af[4], bf[2];
#pragma unroll
    for (int f = 0; f < 4; ++f) {
      const int row = wr * 64 + f * 16 + l15;
      v8us lo = *(const v8us*)(a_lds + row * ROWP + half * 8);
      v8us hi = *(const v8us*)(a_lds + row * ROWP + 16 + half * 8);
      af[f] = mk_frag(lo, hi);
    }
#pragma unroll
    for (int j = 0; j < 2; ++j) {
      const int col = wc * 32 + j * 16 + l15;
      v8us lo = *(const v8us*)(b_lds + col * ROWP + half * 8);
      v8us hi = *(const v8us*)(b_lds + col * ROWP + 16 + half * 8);
      bf[j] = mk_frag(lo, hi);
    }
#pragma unroll
    for (int f = 0; f < 4; ++f)
#pragma unroll
      for (int j = 0; j < 2; ++j) acc[f][j] = wmma_bf16(af[f], bf[j], acc[f][j]);
    __syncthreads();  // all readers done before this buffer is re-staged
  }

#pragma unroll
  for (int f = 0; f < 4; ++f)
#pragma unroll
    for (int j = 0; j < 2; ++j) {
      const int r0 = bm * 256 + wr * 64 + f * 16 + half * 8;
      const int c0 = bn * 64 + wc * 32 + j * 16 + l15;
#pragma unroll
      for (int i = 0; i < 8; ++i)
        Ob[(size_t)(r0 + i) * C + c0] = f32_to_bf16(acc[f][j][i]);
    }
}

// ---------------------------------------------------------------------------
// GEMM2: out(65536x256 f32) = relu(AH(65536x256 bf16) @ W^T + bias)
// ---------------------------------------------------------------------------
__global__ __launch_bounds__(256) void gcn_linear_relu_bf16(
    const unsigned short* __restrict__ AH, const unsigned short* __restrict__ Wb,
    const float* __restrict__ bias, float* __restrict__ out) {
  constexpr int Kd = 256, E = 256;
  constexpr int CT = E / 32;

  const int wave = threadIdx.x >> 5;
  const int lane = threadIdx.x & 31;
  const int tile = blockIdx.x * 8 + wave;
  const int tm   = tile / CT;
  const int tn   = tile % CT;
  const int half = lane >> 4;
  const int l15  = lane & 15;

  const unsigned short* pa[4];
#pragma unroll
  for (int f = 0; f < 4; ++f)
    pa[f] = AH + (size_t)(tm * 64 + f * 16 + l15) * Kd + half * 8;
  const unsigned short* pw[2];
#pragma unroll
  for (int j = 0; j < 2; ++j)
    pw[j] = Wb + (size_t)(tn * 32 + j * 16 + l15) * Kd + half * 8;

  v8f acc[4][2] = {};
#pragma unroll
  for (int k0 = 0; k0 < Kd; k0 += 32) {
    v16bf af[4], bf[2];
#pragma unroll
    for (int f = 0; f < 4; ++f)
      af[f] = mk_frag(*(const v8us*)(pa[f] + k0), *(const v8us*)(pa[f] + k0 + 16));
#pragma unroll
    for (int j = 0; j < 2; ++j)
      bf[j] = mk_frag(*(const v8us*)(pw[j] + k0), *(const v8us*)(pw[j] + k0 + 16));
#pragma unroll
    for (int f = 0; f < 4; ++f)
#pragma unroll
      for (int j = 0; j < 2; ++j) acc[f][j] = wmma_bf16(af[f], bf[j], acc[f][j]);
  }

#pragma unroll
  for (int j = 0; j < 2; ++j) {
    const int e0 = tn * 32 + j * 16 + l15;
    const float bi = bias[e0];
#pragma unroll
    for (int f = 0; f < 4; ++f) {
      const int r0 = tm * 64 + f * 16 + half * 8;
#pragma unroll
      for (int i = 0; i < 8; ++i)
        out[(size_t)(r0 + i) * E + e0] = fmaxf(acc[f][j][i] + bi, 0.0f);
    }
  }
}

extern "C" void kernel_launch(void* const* d_in, const int* in_sizes, int n_in,
                              void* d_out, int out_size, void* d_ws,
                              size_t ws_size, hipStream_t stream) {
  const float* prop_state = (const float*)d_in[0];  // (8,2048,4,256)
  const float* A          = (const float*)d_in[1];  // (8,2048,2048)
  const float* W          = (const float*)d_in[2];  // (256,256)
  const float* bias       = (const float*)d_in[3];  // (256,)
  float* out = (float*)d_out;

  // Workspace (bf16): A 64MiB | Ht 32MiB | AH 32MiB | W 128KiB
  char* ws = (char*)d_ws;
  unsigned short* Abf  = (unsigned short*)ws;
  unsigned short* Htbf = (unsigned short*)(ws + (size_t)67108864);
  unsigned short* AHbf = (unsigned short*)(ws + (size_t)100663296);
  unsigned short* Wbf  = (unsigned short*)(ws + (size_t)134217728);

  cvt_bf16<<<16384, 256, 0, stream>>>(A, Abf);
  cvt_bf16<<<32, 256, 0, stream>>>(W, Wbf);
  transpose_h_bf16<<<4096, 256, 0, stream>>>(prop_state, Htbf);
  // GEMM1: 8 batches * 128 block-tiles = 1024 blocks x 256 threads
  gcn_aggregate_bf16<<<1024, 256, 0, stream>>>(Abf, Htbf, AHbf);
  // GEMM2: 8192 waves -> 1024 blocks x 8 waves
  gcn_linear_relu_bf16<<<1024, 256, 0, stream>>>(AHbf, Wbf, bias, out);
}
